// MultiHeadCredibilityAttention_4733053960678
// MI455X (gfx1250) — matrix operand order repaired and checked by reading, compiled, not run on hardware
//
#include <hip/hip_runtime.h>

// ---------------------------------------------------------------------------
// MI455X / gfx1250 multi-head attention
//   - bf16 WMMA (v_wmma_f32_16x16x32_bf16) for all matmuls
//   - TDM (tensor_load_to_lds + s_wait_tensorcnt) double-buffered K/V staging
// ---------------------------------------------------------------------------

typedef __attribute__((ext_vector_type(16))) __bf16       v16bf;
typedef __attribute__((ext_vector_type(8)))  float        v8f;
typedef __attribute__((ext_vector_type(4)))  unsigned int u32x4;
typedef __attribute__((ext_vector_type(8)))  int          i32x8;
typedef __attribute__((ext_vector_type(4)))  int          i32x4;

union FragBF {
    v16bf          bf;
    u32x4          u4[2];
    unsigned short u16[16];
};

// Problem constants (match reference)
#define BB   4
#define SS   2048
#define DD   256
#define HH   8
#define HD   32
#define MM   (BB * SS)          // 8192 rows total
#define NWAVE 8                 // waves per attention block

__device__ __forceinline__ unsigned short f2bf(float f) {
    unsigned int u = __float_as_uint(f);
    u += 0x7FFFu + ((u >> 16) & 1u);      // round-to-nearest-even
    return (unsigned short)(u >> 16);
}

__device__ __forceinline__ v8f wmma_bf16(const FragBF& a, const FragBF& b, v8f c) {
    return __builtin_amdgcn_wmma_f32_16x16x32_bf16(
        /*neg_a=*/false, a.bf, /*neg_b=*/false, b.bf,
        /*c_mod=*/(short)0, c, /*reuse_a=*/false, /*reuse_b=*/false);
}

// ---------------------------------------------------------------------------
// TDM: issue a 32x32 bf16 tile load Global -> LDS (one D#, 2D, in-bounds).
// D# per CDNA5 ISA ch.8: group0 = {count, lds_addr, global_addr, type=2},
// group1 = {data_size=2B, tensor_dim0=32, tensor_dim1=32, tile 32x32,
//           dim0_stride=32}. Groups 2/3 unused (2D tensor) -> zeros.
// Toolchain here wants the 6-arg builtin (clang-23 / therock-10.0 headers).
// ---------------------------------------------------------------------------
#if __has_builtin(__builtin_amdgcn_tensor_load_to_lds)
#define HAVE_TDM 1
__device__ __forceinline__ void tdm_load_tile32x32(const unsigned short* gsrc,
                                                   unsigned lds_off) {
    const unsigned long long ga = (unsigned long long)(uintptr_t)gsrc;
    u32x4 g0;
    g0[0] = 1u;                                        // count=1, user mode
    g0[1] = lds_off;                                   // lds_addr
    g0[2] = (unsigned)(ga & 0xFFFFFFFFu);              // global_addr[31:0]
    g0[3] = (unsigned)((ga >> 32) & 0x01FFFFFFu)       // global_addr[56:32]
          | (2u << 30);                                // type = 2 ("image")
    i32x8 g1;
    g1[0] = (int)(1u << 16);        // wg_mask=0, data_size=1 (2 bytes)
    g1[1] = (int)(32u << 16);       // tensor_dim0 = 32   ([79:48] low half)
    g1[2] = (int)(32u << 16);       // tensor_dim1 = 32   ([95:80])
    g1[3] = (int)(32u << 16);       // tile_dim0 = 32     ([127:112])
    g1[4] = 32;                     // tile_dim1 = 32     ([143:128])
    g1[5] = 32;                     // tensor_dim0_stride = 32 ([191:160])
    g1[6] = 0;
    g1[7] = 0;
    const i32x4 z4 = {0, 0, 0, 0};
    const i32x8 z8 = {0, 0, 0, 0, 0, 0, 0, 0};
    __builtin_amdgcn_tensor_load_to_lds(g0, g1, z4, z4, z8, 0);
}
#else
#define HAVE_TDM 0
#endif

// ---------------------------------------------------------------------------
// Kernel 0: fp32 -> bf16 conversion
// ---------------------------------------------------------------------------
__global__ void k_cvt_bf16(const float* __restrict__ src,
                           unsigned short* __restrict__ dst, int n) {
    int i = blockIdx.x * blockDim.x + threadIdx.x;
    if (i < n) dst[i] = f2bf(src[i]);
}

// ---------------------------------------------------------------------------
// Kernel 1: Y = x @ W^T + b, output bf16 in (B,H,S,Hd) layout.
// grid = (M/16, D/64), block = 32 (one wave). Each wave: 16x64 tile.
// ---------------------------------------------------------------------------
__global__ void k_qkv_gemm(const unsigned short* __restrict__ xb,   // (M, D) bf16
                           const unsigned short* __restrict__ Wb,   // (D, D) bf16
                           const float* __restrict__ bias,          // (D)
                           unsigned short* __restrict__ outH) {     // (B,H,S,Hd) bf16
    const int lane = threadIdx.x;
    const int nl   = lane & 15;
    const int half = lane >> 4;
    const int m0   = blockIdx.x * 16;
    const int n0   = blockIdx.y * 64;

    v8f acc[4] = {v8f{}, v8f{}, v8f{}, v8f{}};

    const unsigned short* arow = xb + (size_t)(m0 + nl) * DD;
    for (int k0 = 0; k0 < DD; k0 += 32) {
        FragBF a;
        a.u4[0] = *(const u32x4*)(arow + k0 + half * 16);
        a.u4[1] = *(const u32x4*)(arow + k0 + half * 16 + 8);
#pragma unroll
        for (int t = 0; t < 4; ++t) {
            const unsigned short* brow = Wb + (size_t)(n0 + t * 16 + nl) * DD + k0 + half * 16;
            FragBF b;
            b.u4[0] = *(const u32x4*)(brow);
            b.u4[1] = *(const u32x4*)(brow + 8);
            acc[t] = wmma_bf16(a, b, acc[t]);
        }
    }

#pragma unroll
    for (int t = 0; t < 4; ++t) {
        const int   ng = n0 + t * 16 + nl;
        const float bv = bias[ng];
        const int   h  = ng >> 5;
        const int   hd = ng & 31;
#pragma unroll
        for (int r = 0; r < 8; ++r) {
            const int mg = m0 + r + half * 8;
            const int bi = mg >> 11;
            const int s  = mg & (SS - 1);
            outH[(((size_t)(bi * HH + h) * SS + s) << 5) + hd] = f2bf(acc[t][r] + bv);
        }
    }
}

// ---------------------------------------------------------------------------
// Kernel 2: flash attention. block = 256 (8 waves), grid = (S/128, H, B).
// Wave 0 stages 32x32 K/V tiles into LDS via TDM, double-buffered; all 8
// waves (8 x 16 query rows) consume each tile. Online softmax, P via LDS
// transpose staging (per-wave slice, same-wave DS is in-order).
// ---------------------------------------------------------------------------
__global__ void __launch_bounds__(256)
k_attn(const unsigned short* __restrict__ Qb,   // (B,H,S,Hd) bf16
       const unsigned short* __restrict__ Kb,
       const unsigned short* __restrict__ Vb,
       const float* __restrict__ head_scale,    // (H)
       unsigned short* __restrict__ Ctx) {      // (B,S,D) bf16
    __shared__ unsigned short Kt[2][32 * 32];           // 2 x 2KB
    __shared__ unsigned short Vt[2][32 * 32];           // 2 x 2KB
    __shared__ unsigned short Plds[NWAVE][16][32];      // 8KB

    const int tid  = threadIdx.x;
    const int wave = tid >> 5;
    const int lane = tid & 31;
    const int nl   = lane & 15;
    const int half = lane >> 4;
    const int h    = blockIdx.y;
    const int bi   = blockIdx.z;
    const int m0   = (blockIdx.x * NWAVE + wave) * 16;

    const size_t hb = (size_t)(bi * HH + h) * SS * HD;
    const unsigned short* Qh = Qb + hb;
    const unsigned short* Kh = Kb + hb;
    const unsigned short* Vh = Vb + hb;

    // Q A-fragment (16 rows x 32 head dims), reused across all key tiles
    FragBF aq;
    {
        const unsigned short* qrow = Qh + (size_t)(m0 + nl) * HD;
        aq.u4[0] = *(const u32x4*)(qrow + half * 16);
        aq.u4[1] = *(const u32x4*)(qrow + half * 16 + 8);
    }

    const float sm_scale = 0.17677669529663687f;  // 1/sqrt(Hd)
    float mrun[8], lrun[8];
    v8f o0 = {}, o1 = {};
#pragma unroll
    for (int r = 0; r < 8; ++r) { mrun[r] = -1e30f; lrun[r] = 0.0f; }

    const unsigned kt_off[2] = { (unsigned)(uintptr_t)&Kt[0][0],
                                 (unsigned)(uintptr_t)&Kt[1][0] };
    const unsigned vt_off[2] = { (unsigned)(uintptr_t)&Vt[0][0],
                                 (unsigned)(uintptr_t)&Vt[1][0] };

    const int NT = SS / 32;    // 64 key tiles

#if HAVE_TDM
    if (wave == 0) {           // prologue: stage tile 0 into buffer 0
        tdm_load_tile32x32(Kh, kt_off[0]);
        tdm_load_tile32x32(Vh, vt_off[0]);
    }
#else
    {   // fallback staging: 256 threads x 16B covers K tile + V tile (4KB)
        const unsigned short* src = (tid < 128) ? (Kh + tid * 8) : (Vh + (tid - 128) * 8);
        unsigned short* dst = (tid < 128) ? (&Kt[0][0] + tid * 8) : (&Vt[0][0] + (tid - 128) * 8);
        *(u32x4*)dst = *(const u32x4*)src;
    }
#endif

    for (int j = 0; j < NT; ++j) {
        const int buf = j & 1;

#if HAVE_TDM
        if (wave == 0) __builtin_amdgcn_s_wait_tensorcnt(0);   // tile j landed
#endif
        __syncthreads();                                        // publish tile j

#if HAVE_TDM
        if (j + 1 < NT && wave == 0) {   // prefetch tile j+1 (overlaps compute)
            const size_t nk = (size_t)(j + 1) * 32 * HD;
            tdm_load_tile32x32(Kh + nk, kt_off[buf ^ 1]);
            tdm_load_tile32x32(Vh + nk, vt_off[buf ^ 1]);
        }
#endif

        // --- K^T B-fragments from LDS (contiguous 32B per lane) ---
        const unsigned short* kb = &Kt[buf][0];
        FragBF bk0, bk1;
        bk0.u4[0] = *(const u32x4*)(kb + nl * 32 + half * 16);
        bk0.u4[1] = *(const u32x4*)(kb + nl * 32 + half * 16 + 8);
        bk1.u4[0] = *(const u32x4*)(kb + (16 + nl) * 32 + half * 16);
        bk1.u4[1] = *(const u32x4*)(kb + (16 + nl) * 32 + half * 16 + 8);

        v8f s0 = {}, s1 = {};
        s0 = wmma_bf16(aq, bk0, s0);       // scores, keys j*32 .. +15
        s1 = wmma_bf16(aq, bk1, s1);       // scores, keys j*32+16 .. +31

        // --- online softmax (row = r + 8*half, col = nl in-half) ---
        float p0[8], p1[8];
#pragma unroll
        for (int r = 0; r < 8; ++r) {
            float x0 = s0[r] * sm_scale;
            float x1 = s1[r] * sm_scale;
            float mx = fmaxf(x0, x1);
            mx = fmaxf(mx, __shfl_xor(mx, 1, 32));
            mx = fmaxf(mx, __shfl_xor(mx, 2, 32));
            mx = fmaxf(mx, __shfl_xor(mx, 4, 32));
            mx = fmaxf(mx, __shfl_xor(mx, 8, 32));
            const float mnew = fmaxf(mrun[r], mx);
            const float corr = __expf(mrun[r] - mnew);
            x0 = __expf(x0 - mnew);
            x1 = __expf(x1 - mnew);
            float ps = x0 + x1;
            ps += __shfl_xor(ps, 1, 32);
            ps += __shfl_xor(ps, 2, 32);
            ps += __shfl_xor(ps, 4, 32);
            ps += __shfl_xor(ps, 8, 32);
            lrun[r] = lrun[r] * corr + ps;
            mrun[r] = mnew;
            o0[r] *= corr;
            o1[r] *= corr;
            p0[r] = x0;
            p1[r] = x1;
        }

        // --- P: C-layout -> per-wave LDS slice -> A-fragment ---
#pragma unroll
        for (int r = 0; r < 8; ++r) {
            Plds[wave][r + half * 8][nl]      = f2bf(p0[r]);
            Plds[wave][r + half * 8][16 + nl] = f2bf(p1[r]);
        }
        FragBF ap;
        ap.u4[0] = *(const u32x4*)(&Plds[wave][nl][half * 16]);
        ap.u4[1] = *(const u32x4*)(&Plds[wave][nl][half * 16 + 8]);

        // --- V B-fragments from LDS tile ---
        const unsigned short* vb = &Vt[buf][0];
        FragBF bv0, bv1;
#pragma unroll
        for (int i = 0; i < 16; ++i) {
            bv0.u16[i] = vb[(half * 16 + i) * 32 + nl];
            bv1.u16[i] = vb[(half * 16 + i) * 32 + 16 + nl];
        }

        o0 = wmma_bf16(ap, bv0, o0);
        o1 = wmma_bf16(ap, bv1, o1);

#if !HAVE_TDM
        __syncthreads();
        if (j + 1 < NT) {
            const size_t nk = (size_t)(j + 1) * 32 * HD;
            const unsigned short* src = (tid < 128) ? (Kh + nk + tid * 8)
                                                    : (Vh + nk + (tid - 128) * 8);
            unsigned short* dst = (tid < 128) ? (&Kt[buf ^ 1][0] + tid * 8)
                                              : (&Vt[buf ^ 1][0] + (tid - 128) * 8);
            *(u32x4*)dst = *(const u32x4*)src;
        }
#endif
    }

    const float hs = head_scale[h];
#pragma unroll
    for (int r = 0; r < 8; ++r) {
        const int   mg  = m0 + r + half * 8;
        const float inv = hs / lrun[r];
        unsigned short* crow = Ctx + (size_t)(bi * SS + mg) * DD + h * HD;
        crow[nl]      = f2bf(o0[r] * inv);
        crow[16 + nl] = f2bf(o1[r] * inv);
    }
}

// ---------------------------------------------------------------------------
// Kernel 3: out = Ctx @ Wo^T + bo, fp32 output (B,S,D).
// ---------------------------------------------------------------------------
__global__ void k_out_gemm(const unsigned short* __restrict__ Ctxb,  // (M, D) bf16
                           const unsigned short* __restrict__ Wob,   // (D, D) bf16
                           const float* __restrict__ bo,
                           float* __restrict__ out) {                // (M, D) f32
    const int lane = threadIdx.x;
    const int nl   = lane & 15;
    const int half = lane >> 4;
    const int m0   = blockIdx.x * 16;
    const int n0   = blockIdx.y * 64;

    v8f acc[4] = {v8f{}, v8f{}, v8f{}, v8f{}};

    const unsigned short* arow = Ctxb + (size_t)(m0 + nl) * DD;
    for (int k0 = 0; k0 < DD; k0 += 32) {
        FragBF a;
        a.u4[0] = *(const u32x4*)(arow + k0 + half * 16);
        a.u4[1] = *(const u32x4*)(arow + k0 + half * 16 + 8);
#pragma unroll
        for (int t = 0; t < 4; ++t) {
            const unsigned short* brow = Wob + (size_t)(n0 + t * 16 + nl) * DD + k0 + half * 16;
            FragBF b;
            b.u4[0] = *(const u32x4*)(brow);
            b.u4[1] = *(const u32x4*)(brow + 8);
            acc[t] = wmma_bf16(a, b, acc[t]);
        }
    }

#pragma unroll
    for (int t = 0; t < 4; ++t) {
        const int   ng = n0 + t * 16 + nl;
        const float bv = bo[ng];
#pragma unroll
        for (int r = 0; r < 8; ++r) {
            const int mg = m0 + r + half * 8;
            out[(size_t)mg * DD + ng] = acc[t][r] + bv;
        }
    }
}

// ---------------------------------------------------------------------------
// Launch
// ---------------------------------------------------------------------------
extern "C" void kernel_launch(void* const* d_in, const int* in_sizes, int n_in,
                              void* d_out, int out_size, void* d_ws, size_t ws_size,
                              hipStream_t stream) {
    const float* x   = (const float*)d_in[0];
    const float* Wq  = (const float*)d_in[1];
    const float* bq  = (const float*)d_in[2];
    const float* Wk  = (const float*)d_in[3];
    const float* bk  = (const float*)d_in[4];
    const float* Wv  = (const float*)d_in[5];
    const float* bv  = (const float*)d_in[6];
    const float* Wo  = (const float*)d_in[7];
    const float* bo  = (const float*)d_in[8];
    const float* hs  = (const float*)d_in[9];
    float* out = (float*)d_out;

    char* ws = (char*)d_ws;
    unsigned short* xb  = (unsigned short*)(ws);                     //  4 MB  (M*D)
    unsigned short* Wqb = (unsigned short*)(ws + (4u << 20));        //  128KB each
    unsigned short* Wkb = Wqb + DD * DD;
    unsigned short* Wvb = Wkb + DD * DD;
    unsigned short* Wob = Wvb + DD * DD;
    unsigned short* Qb  = (unsigned short*)(ws + (5u << 20));        //  4 MB
    unsigned short* Kb  = Qb + (size_t)MM * DD;                      //  4 MB
    unsigned short* Vb  = Kb + (size_t)MM * DD;                      //  4 MB
    unsigned short* Ctx = Vb + (size_t)MM * DD;                      //  4 MB

    const int NX = MM * DD;     // 2,097,152
    const int NW = DD * DD;     // 65,536

    k_cvt_bf16<<<dim3((NX + 255) / 256), dim3(256), 0, stream>>>(x,  xb,  NX);
    k_cvt_bf16<<<dim3((NW + 255) / 256), dim3(256), 0, stream>>>(Wq, Wqb, NW);
    k_cvt_bf16<<<dim3((NW + 255) / 256), dim3(256), 0, stream>>>(Wk, Wkb, NW);
    k_cvt_bf16<<<dim3((NW + 255) / 256), dim3(256), 0, stream>>>(Wv, Wvb, NW);
    k_cvt_bf16<<<dim3((NW + 255) / 256), dim3(256), 0, stream>>>(Wo, Wob, NW);

    dim3 ggrid(MM / 16, DD / 64);          // (512, 4)
    k_qkv_gemm<<<ggrid, dim3(32), 0, stream>>>(xb, Wqb, bq, Qb);
    k_qkv_gemm<<<ggrid, dim3(32), 0, stream>>>(xb, Wkb, bk, Kb);
    k_qkv_gemm<<<ggrid, dim3(32), 0, stream>>>(xb, Wvb, bv, Vb);

    k_attn<<<dim3(SS / (16 * NWAVE), HH, BB), dim3(256), 0, stream>>>(Qb, Kb, Vb, hs, Ctx);

    k_out_gemm<<<ggrid, dim3(32), 0, stream>>>(Ctx, Wob, bo, out);
}
